// SparseMOE_44667659878536
// MI455X (gfx1250) — matrix operand order repaired and checked by reading, not compiled, not run on hardware
//
#include <hip/hip_runtime.h>
#include <cmath>

// Sparse MoE forward for MI455X (gfx1250, wave32).
//   Router (1 wave/token) -> deterministic ballot-compacted expert lists ->
//   fc1 GEMM (bf16 WMMA 16x16x32, f32 acc, exact erf-GELU) -> packed bf16 h ->
//   fc2 GEMM (bf16 WMMA, A-tile via TDM tensor_load_to_lds) -> combine-weighted
//   f32 atomic scatter into zeroed out.
// Compute-bound: ~275 GFLOP useful work vs ~22 us of HBM traffic at 23.3 TB/s.
// K-chunk = 64: 8 v_wmma per LDS stage per wave between barriers.

#define NEXP 8
#define DIM  1024
#define HID  4096
#define NTOK 8192   // B*S = 4*2048
#define TOPK 2

typedef __attribute__((ext_vector_type(16))) __bf16 v16bf;
typedef __attribute__((ext_vector_type(8)))  __bf16 v8bf;
typedef __attribute__((ext_vector_type(8)))  float  v8f;
typedef __attribute__((ext_vector_type(4)))  unsigned int v4u;
typedef __attribute__((ext_vector_type(4)))  int v4i;
typedef __attribute__((ext_vector_type(8)))  int v8i;

#if defined(__HIP_DEVICE_COMPILE__) && \
    __has_builtin(__builtin_amdgcn_tensor_load_to_lds) && \
    __has_builtin(__builtin_amdgcn_s_wait_tensorcnt)
#define USE_TDM 1
#else
#define USE_TDM 0
#endif

// ---------------------------------------------------------------- router ----
__global__ __launch_bounds__(256) void moe_router(const float* __restrict__ x,
                                                  const float* __restrict__ Wr,
                                                  float* __restrict__ combine) {
  const int wave = threadIdx.x >> 5;
  const int lane = threadIdx.x & 31;
  const int t = blockIdx.x * 8 + wave;
  float p[NEXP];
#pragma unroll
  for (int e = 0; e < NEXP; ++e) p[e] = 0.f;
  const float* xr = x + (size_t)t * DIM;
  for (int i = lane; i < DIM; i += 32) {
    float xi = xr[i];
#pragma unroll
    for (int e = 0; e < NEXP; ++e) p[e] += xi * Wr[e * DIM + i];
  }
#pragma unroll
  for (int e = 0; e < NEXP; ++e) {
#pragma unroll
    for (int off = 16; off > 0; off >>= 1) p[e] += __shfl_xor(p[e], off, 32);
  }
  if (lane == 0) {
    float mx = p[0];
#pragma unroll
    for (int e = 1; e < NEXP; ++e) mx = fmaxf(mx, p[e]);
    float s = 0.f;
#pragma unroll
    for (int e = 0; e < NEXP; ++e) { p[e] = __expf(p[e] - mx); s += p[e]; }
    float inv = 1.f / s;
#pragma unroll
    for (int e = 0; e < NEXP; ++e) p[e] *= inv;
    int i1 = 0;
    for (int e = 1; e < NEXP; ++e) if (p[e] > p[i1]) i1 = e;  // first-max (ties -> lower idx)
    int i2 = -1;
    for (int e = 0; e < NEXP; ++e) {
      if (e == i1) continue;
      if (i2 < 0 || p[e] > p[i2]) i2 = e;
    }
#pragma unroll
    for (int e = 0; e < NEXP; ++e)
      combine[(size_t)t * NEXP + e] = (e == i1) ? p[i1] : (e == i2 ? p[i2] : 0.f);
  }
}

// ----------------------------------------------- deterministic list build ----
__global__ __launch_bounds__(32) void moe_build_lists(const float* __restrict__ combine,
                                                      int* __restrict__ counts,
                                                      int* __restrict__ offsets,
                                                      int* __restrict__ list) {
  const int lane = threadIdx.x;  // 32 threads, wave32
  int total = 0;
  for (int e = 0; e < NEXP; ++e) {
    if (lane == 0) offsets[e] = total;
    int cnt = 0;
    for (int base = 0; base < NTOK; base += 32) {
      int t = base + lane;
      bool sel = combine[(size_t)t * NEXP + e] > 0.f;
      unsigned long long m = __ballot(sel);
      if (sel) {
        int pos = __popcll(m & ((1ull << lane) - 1ull));
        list[total + cnt + pos] = t;
      }
      cnt += __popcll(m);
    }
    if (lane == 0) counts[e] = cnt;
    total += cnt;
  }
  if (lane == 0) offsets[NEXP] = total;
}

// ------------------------------------------------------------- fc1 (GEMM) ----
// C[64 x 128] tile per workgroup; 8 waves as 2(M) x 4(N); each wave 32x32 via
// 2x2 WMMA, K-chunk 64 -> 8 v_wmma_f32_16x16x32_bf16 per stage per wave.
__global__ __launch_bounds__(256) void moe_fc1(const float* __restrict__ x,
                                               const float* __restrict__ W1,
                                               const float* __restrict__ b1,
                                               const int* __restrict__ counts,
                                               const int* __restrict__ offsets,
                                               const int* __restrict__ list,
                                               __bf16* __restrict__ h) {
  const int e = blockIdx.z;
  const int cnt = counts[e];
  const int m0 = blockIdx.y * 64;
  if (m0 >= cnt) return;
  const int n0 = blockIdx.x * 128;
  const int off = offsets[e];
  const float* W1e = W1 + (size_t)e * DIM * HID;

  __shared__ alignas(64) __bf16 As[64][64];
  __shared__ alignas(64) __bf16 Bs[128][64];  // transposed: [n][k]
  __shared__ int rowTok[64];

  const int tid = threadIdx.x;
  const int wave = tid >> 5, lane = tid & 31;
  const int wm = wave >> 2, wn = wave & 3;

  if (tid < 64) {
    int r = m0 + tid;
    rowTok[tid] = list[off + (r < cnt ? r : 0)];
  }
  __syncthreads();

  v8f acc[2][2];
#pragma unroll
  for (int mi = 0; mi < 2; ++mi)
#pragma unroll
    for (int ni = 0; ni < 2; ++ni) {
      v8f z = {0.f, 0.f, 0.f, 0.f, 0.f, 0.f, 0.f, 0.f};
      acc[mi][ni] = z;
    }

  // A loader: 4 threads/row, 16 consecutive f32 each.  B loader: 1 k-row,
  // 32 consecutive n each, stored transposed.
  const int la_row = tid >> 2;
  const int la_col = (tid & 3) * 16;
  const int lb_k   = tid >> 2;        // 0..63
  const int lb_n   = (tid & 3) * 32;  // 0,32,64,96
  const float* xrow = x + (size_t)rowTok[la_row] * DIM;  // hoisted gather base

  // Hoisted fragment bases (A/B: contiguous k = 16*(lane/16) + s).
  const int arow = wm * 32 + (lane & 15);
  const int akb = (lane >> 4) * 16;   // A: lane<16 -> K{0..15}, else K{16..31}
  const int bn = wn * 32 + (lane & 15);
  const int bkb = (lane >> 4) * 16;   // B: lane column, contiguous 16-K run per half-wave

  for (int k0 = 0; k0 < DIM; k0 += 64) {
    {  // A tile: gathered token rows, fp32 -> bf16
      const float* src = xrow + k0 + la_col;
#pragma unroll
      for (int j = 0; j < 16; ++j) As[la_row][la_col + j] = (__bf16)src[j];
    }
    {  // B tile: W1[e][k][n] -> Bs[n][k], fp32 -> bf16
      const float* src = W1e + (size_t)(k0 + lb_k) * HID + n0 + lb_n;
#pragma unroll
      for (int j = 0; j < 32; ++j) Bs[lb_n + j][lb_k] = (__bf16)src[j];
      if (k0 + 64 < DIM)  // global_prefetch_b8 next K-chunk of weights
        __builtin_prefetch(W1e + (size_t)(k0 + 64 + lb_k) * HID + n0 + lb_n, 0, 3);
    }
    __syncthreads();

#pragma unroll
    for (int kc = 0; kc < 2; ++kc) {
      v16bf a[2], b[2];
#pragma unroll
      for (int mi = 0; mi < 2; ++mi) {
        const __bf16* pa = &As[arow + mi * 16][kc * 32];
        v8bf lo = *(const v8bf*)(pa + akb);
        v8bf hi = *(const v8bf*)(pa + akb + 8);
        a[mi] = __builtin_shufflevector(lo, hi, 0, 1, 2, 3, 4, 5, 6, 7, 8, 9, 10, 11, 12, 13, 14, 15);
      }
#pragma unroll
      for (int ni = 0; ni < 2; ++ni)
        b[ni] = *(const v16bf*)&Bs[bn + ni * 16][kc * 32 + bkb];
#pragma unroll
      for (int mi = 0; mi < 2; ++mi)
#pragma unroll
        for (int ni = 0; ni < 2; ++ni)
          acc[mi][ni] = __builtin_amdgcn_wmma_f32_16x16x32_bf16(
              false, a[mi], false, b[ni], (short)0, acc[mi][ni], false, false);
    }
    __syncthreads();
  }

  // Epilogue: bias + exact GELU, store packed bf16 h.
  const int mbase = wm * 32 + ((lane >> 4) << 3);  // C: VGPR i -> M=i (+8 for hi half-wave)
  const int nloc = lane & 15;
#pragma unroll
  for (int mi = 0; mi < 2; ++mi) {
#pragma unroll
    for (int ni = 0; ni < 2; ++ni) {
#pragma unroll
      for (int i = 0; i < 8; ++i) {
        int r = m0 + mbase + mi * 16 + i;
        if (r < cnt) {
          int n = n0 + wn * 32 + ni * 16 + nloc;
          float v = acc[mi][ni][i] + b1[e * HID + n];
          v = 0.5f * v * (1.f + erff(v * 0.70710678118654752440f));
          h[(size_t)(off + r) * HID + n] = (__bf16)v;
        }
      }
    }
  }
}

// ------------------------------------------------------------- fc2 (GEMM) ----
// A-tile (packed bf16 h rows, plain 2D strided copy) moved by the Tensor Data
// Mover: wave 0 issues tensor_load_to_lds with a D# describing a 64x64 2B
// tile, tensor_dim1 = remaining rows (hardware zero-fills the ragged last
// tile), then s_wait_tensorcnt + barrier.
__global__ __launch_bounds__(256) void moe_fc2(const __bf16* __restrict__ h,
                                               const float* __restrict__ W2,
                                               const float* __restrict__ b2,
                                               const int* __restrict__ counts,
                                               const int* __restrict__ offsets,
                                               const int* __restrict__ list,
                                               const float* __restrict__ combine,
                                               float* __restrict__ out) {
  const int e = blockIdx.z;
  const int cnt = counts[e];
  const int m0 = blockIdx.y * 64;
  if (m0 >= cnt) return;
  const int n0 = blockIdx.x * 128;
  const int off = offsets[e];
  const float* W2e = W2 + (size_t)e * HID * DIM;

  __shared__ alignas(64) __bf16 As[64][64];
  __shared__ alignas(64) __bf16 Bs[128][64];  // transposed [n][k]
  __shared__ int rowTok[64];
  __shared__ float rowW[64];

  const int tid = threadIdx.x;
  const int wave = tid >> 5, lane = tid & 31;
  const int wm = wave >> 2, wn = wave & 3;

  if (tid < 64) {
    int r = m0 + tid;
    int t = list[off + (r < cnt ? r : 0)];
    rowTok[tid] = t;
    rowW[tid] = combine[(size_t)t * NEXP + e];
  }
  __syncthreads();

  v8f acc[2][2];
#pragma unroll
  for (int mi = 0; mi < 2; ++mi)
#pragma unroll
    for (int ni = 0; ni < 2; ++ni) {
      v8f z = {0.f, 0.f, 0.f, 0.f, 0.f, 0.f, 0.f, 0.f};
      acc[mi][ni] = z;
    }

  const int lb_k = tid >> 2;
  const int lb_n = (tid & 3) * 32;
#if !USE_TDM
  const int la_row = tid >> 2;
  const int la_col = (tid & 3) * 16;
  const int arow_cl = (m0 + la_row < cnt) ? (m0 + la_row) : 0;
  const __bf16* hrow = h + (size_t)(off + arow_cl) * HID;
#else
  // D# pieces that are loop-invariant.
  const unsigned int ldsA = (unsigned int)(size_t)&As[0][0];
  unsigned int rem = (unsigned int)(cnt - m0);
  if (rem > 64u) rem = 64u;
  const unsigned long long gbase =
      (unsigned long long)(size_t)(h + (size_t)(off + m0) * HID);
#endif

  const int arow = wm * 32 + (lane & 15);
  const int akb = (lane >> 4) * 16;
  const int bn = wn * 32 + (lane & 15);
  const int bkb = (lane >> 4) * 16;

  for (int k0 = 0; k0 < HID; k0 += 64) {
#if USE_TDM
    if (wave == 0) {
      const unsigned long long ga = gbase + (unsigned long long)k0 * 2u;
      v4u g0 = {1u,                                   // count=1 (valid D#)
                ldsA,                                 // lds_addr
                (unsigned int)ga,                     // global_addr[31:0]
                ((unsigned int)(ga >> 32) & 0x01FFFFFFu) | 0x80000000u};  // ga[56:32] | type=2
      v8i g1 = {(int)(1u << 16),                      // data_size=1 (2 bytes)
                (int)(64u << 16),                     // tensor_dim0 = 64 (lo16 in bits 63:48)
                (int)((rem & 0xFFFFu) << 16),         // dim0 hi16=0 | tensor_dim1 lo16
                (int)(64u << 16),                     // dim1 hi16=0 | tile_dim0 = 64
                (int)64,                              // tile_dim1 = 64, tile_dim2 = 0
                (int)HID,                             // tensor_dim0_stride lo32 = 4096
                0,                                    // stride0 hi16 | stride1 lo16
                0};                                   // stride1 hi32
      v4i g2 = {0, 0, 0, 0};
      v4i g3 = {0, 0, 0, 0};
      v8i g4 = {0, 0, 0, 0, 0, 0, 0, 0};              // 6-arg toolchain extra group (unused)
      __builtin_amdgcn_tensor_load_to_lds(g0, g1, g2, g3, g4, 0);
    }
#else
    *(v8bf*)&As[la_row][la_col] = *(const v8bf*)(hrow + k0 + la_col);
    *(v8bf*)&As[la_row][la_col + 8] = *(const v8bf*)(hrow + k0 + la_col + 8);
#endif
    {  // B tile: W2[e][k][n] -> Bs[n][k], fp32 -> bf16
      const float* src = W2e + (size_t)(k0 + lb_k) * DIM + n0 + lb_n;
#pragma unroll
      for (int j = 0; j < 32; ++j) Bs[lb_n + j][lb_k] = (__bf16)src[j];
      if (k0 + 64 < HID)
        __builtin_prefetch(W2e + (size_t)(k0 + 64 + lb_k) * DIM + n0 + lb_n, 0, 3);
    }
#if USE_TDM
    if (wave == 0) __builtin_amdgcn_s_wait_tensorcnt(0);
#endif
    __syncthreads();

#pragma unroll
    for (int kc = 0; kc < 2; ++kc) {
      v16bf a[2], b[2];
#pragma unroll
      for (int mi = 0; mi < 2; ++mi) {
        const __bf16* pa = &As[arow + mi * 16][kc * 32];
        v8bf lo = *(const v8bf*)(pa + akb);
        v8bf hi = *(const v8bf*)(pa + akb + 8);
        a[mi] = __builtin_shufflevector(lo, hi, 0, 1, 2, 3, 4, 5, 6, 7, 8, 9, 10, 11, 12, 13, 14, 15);
      }
#pragma unroll
      for (int ni = 0; ni < 2; ++ni)
        b[ni] = *(const v16bf*)&Bs[bn + ni * 16][kc * 32 + bkb];
#pragma unroll
      for (int mi = 0; mi < 2; ++mi)
#pragma unroll
        for (int ni = 0; ni < 2; ++ni)
          acc[mi][ni] = __builtin_amdgcn_wmma_f32_16x16x32_bf16(
              false, a[mi], false, b[ni], (short)0, acc[mi][ni], false, false);
    }
    __syncthreads();
  }

  // Epilogue: (y + b2) * combine -> atomic f32 scatter (exactly 2 adds/element
  // onto a zeroed buffer; fp add is commutative -> bitwise deterministic).
  const int mbase = wm * 32 + ((lane >> 4) << 3);
  const int nloc = lane & 15;
#pragma unroll
  for (int mi = 0; mi < 2; ++mi) {
#pragma unroll
    for (int ni = 0; ni < 2; ++ni) {
#pragma unroll
      for (int i = 0; i < 8; ++i) {
        int m = mbase + mi * 16 + i;
        int r = m0 + m;
        if (r < cnt) {
          int n = n0 + wn * 32 + ni * 16 + nloc;
          float v = (acc[mi][ni][i] + b2[e * DIM + n]) * rowW[m];
          atomicAdd(&out[(size_t)rowTok[m] * DIM + n], v);
        }
      }
    }
  }
}

// ---------------------------------------------------------------- launch ----
extern "C" void kernel_launch(void* const* d_in, const int* in_sizes, int n_in,
                              void* d_out, int out_size, void* d_ws, size_t ws_size,
                              hipStream_t stream) {
  const float* x  = (const float*)d_in[0];
  const float* Wr = (const float*)d_in[1];
  const float* W1 = (const float*)d_in[2];
  const float* b1 = (const float*)d_in[3];
  const float* W2 = (const float*)d_in[4];
  const float* b2 = (const float*)d_in[5];
  float* out = (float*)d_out;

  // Workspace layout (needs ~134.5 MB):
  //   [0, 256K)        combine  f32 [NTOK][NEXP]
  //   [256K, +64)      counts   int [NEXP]
  //   [256K+64, +64)   offsets  int [NEXP+1]
  //   [256K+256, +64K) list     int [NTOK*TOPK]  (packed by expert, token order)
  //   [327936, +128M)  h        bf16 [NTOK*TOPK][HID] (packed rows)
  char* ws = (char*)d_ws;
  float* combine = (float*)ws;
  int* counts  = (int*)(ws + 262144);
  int* offsets = (int*)(ws + 262144 + 64);
  int* list    = (int*)(ws + 262144 + 256);
  __bf16* h    = (__bf16*)(ws + 327936);

  (void)hipMemsetAsync(d_out, 0, (size_t)out_size * sizeof(float), stream);
  moe_router<<<NTOK / 8, 256, 0, stream>>>(x, Wr, combine);
  moe_build_lists<<<1, 32, 0, stream>>>(combine, counts, offsets, list);
  moe_fc1<<<dim3(HID / 128, NTOK / 64, NEXP), 256, 0, stream>>>(x, W1, b1, counts, offsets, list, h);
  moe_fc2<<<dim3(DIM / 128, NTOK / 64, NEXP), 256, 0, stream>>>(h, W2, b2, counts, offsets, list,
                                                                combine, out);
}
